// BasicDeconvolutionBlock_4063039062846
// MI455X (gfx1250) — matrix operand order repaired and checked by reading, not compiled, not run on hardware
//
#include <hip/hip_runtime.h>

// ---------------- problem constants (from reference) ----------------
#define K_TAPS      27
#define M_PTS       100000
#define C_CH        64
#define N_OUT_PTS   400000
#define TILES_PER_K (M_PTS / 16)            // 6250 exact
#define ACC_FLOATS  ((size_t)N_OUT_PTS * C_CH)  // 25,600,000 (fits 32-bit offsets)
#define BN_EPS      1e-5f

// padded LDS stride in halves: 72*2 = 144 B -> lane bank step 36, gcd(36,64)=4,
// 16 lanes hit 16 distinct banks (conflict-free), and 144 keeps 16B alignment.
#define LDS_STRIDE  72

typedef __attribute__((ext_vector_type(16))) _Float16 v16h;
typedef __attribute__((ext_vector_type(8)))  _Float16 v8h;
typedef __attribute__((ext_vector_type(8)))  float    v8f;

// ---------------- kernel 1: zero accumulator + stats ----------------
__global__ __launch_bounds__(256) void dz_zero_ws(float4* __restrict__ ws, long n4) {
  long i = (long)blockIdx.x * blockDim.x + threadIdx.x;
  long stride = (long)gridDim.x * blockDim.x;
  float4 z; z.x = 0.f; z.y = 0.f; z.z = 0.f; z.w = 0.f;
  for (; i < n4; i += stride) ws[i] = z;
}

// ---------------- kernel 2: gather -> f16 WMMA GEMM -> scatter-add ----------------
// grid = (ceil(TILES_PER_K/8), K_TAPS), block = 256 (8 wave32)
// wave handles a 16-row tile: [16 x 64] feats  @  [64 x 64] W[k]  -> 16x64, atomically
// added into the f32 accumulator at out_idx rows.
__global__ __launch_bounds__(256) void dz_gather_gemm_scatter(
    const float* __restrict__ x, const float* __restrict__ W,
    const int* __restrict__ in_idx, const int* __restrict__ out_idx,
    float* __restrict__ acc)
{
  __shared__ alignas(16) _Float16 sW[C_CH * LDS_STRIDE];        // W[k]^T: sW[n*72 + kk]
  __shared__ alignas(16) _Float16 sA[8][16 * LDS_STRIDE];       // per-wave feats tile
  __shared__ int sRow[8][16];

  const int t   = threadIdx.x;
  const int w   = t >> 5;        // wave id 0..7
  const int l   = t & 31;        // lane
  const int grp = l >> 4;        // lane group 0/1
  const int lm  = l & 15;
  const int k   = blockIdx.y;

  // --- block-cooperative load of W[k] (64x64 f32), transposed + f16-converted ---
  const float* Wk = W + k * (C_CH * C_CH);
  #pragma unroll
  for (int i = 0; i < 16; ++i) {
    int e  = i * 256 + t;                 // coalesced
    int kk = e >> 6, n = e & 63;
    sW[n * LDS_STRIDE + kk] = (_Float16)Wk[e];
  }

  const int  tile  = blockIdx.x * 8 + w;
  const bool valid = (tile < TILES_PER_K);   // wave-uniform => EXEC all-ones inside

  if (valid) {
    const int m0 = tile * 16;
    // out-row indices for this tile
    if (l < 16) sRow[w][l] = out_idx[k * M_PTS + m0 + l];
    // gather: lane handles row lm, 32 columns starting at 32*grp
    const unsigned irow  = (unsigned)in_idx[k * M_PTS + m0 + lm];
    const int      cbase = grp * 32;
    const float4* xr = (const float4*)(x + irow * (unsigned)C_CH + (unsigned)cbase);
    _Float16* dst = &sA[w][lm * LDS_STRIDE + cbase];
    #pragma unroll
    for (int q = 0; q < 4; ++q) {
      float4 f0 = xr[2 * q], f1 = xr[2 * q + 1];
      v8h h;
      h[0] = (_Float16)f0.x; h[1] = (_Float16)f0.y; h[2] = (_Float16)f0.z; h[3] = (_Float16)f0.w;
      h[4] = (_Float16)f1.x; h[5] = (_Float16)f1.y; h[6] = (_Float16)f1.z; h[7] = (_Float16)f1.w;
      *(v8h*)(dst + 8 * q) = h;            // ds_store_b128, 16B aligned
    }
  }
  __syncthreads();
  if (!valid) return;

  // --- build A fragments per ISA 16-bit 16x32 A layout ---
  // lane group 0: halves 0-7 -> K 0-7,  halves 8-15 -> K 16-23
  // lane group 1: halves 0-7 -> K 8-15, halves 8-15 -> K 24-31
  v16h a0, a1;
  {
    const _Float16* p = &sA[w][lm * LDS_STRIDE + grp * 8];   // K-chunk 0
    *(v8h*)&a0       = *(const v8h*)(p);
    *((v8h*)&a0 + 1) = *(const v8h*)(p + 16);
    const _Float16* q = p + 32;                              // K-chunk 1
    *(v8h*)&a1       = *(const v8h*)(q);
    *((v8h*)&a1 + 1) = *(const v8h*)(q + 16);
  }

  // --- 4 N-tiles x 2 K-chunks of V_WMMA_F32_16X16X32_F16 ---
  v8f cAcc[4];
  #pragma unroll
  for (int j = 0; j < 4; ++j) {
    // B layout: lane = column N (lm), halves 0-7 -> K = kc*32 + grp*8 + h, halves 8-15 -> K = kc*32 + 16 + grp*8 + (h-8)
    const _Float16* bp0 = sW + (16 * j + lm) * LDS_STRIDE + grp * 8;   // kc=0
    const _Float16* bp1 = bp0 + 32;                                     // kc=1
    v16h b0, b1;
    *(v8h*)&b0       = *(const v8h*)(bp0);
    *((v8h*)&b0 + 1) = *(const v8h*)(bp0 + 16);
    *(v8h*)&b1       = *(const v8h*)(bp1);
    *((v8h*)&b1 + 1) = *(const v8h*)(bp1 + 16);
    v8f z = {};
    cAcc[j] = __builtin_amdgcn_wmma_f32_16x16x32_f16(false, a0, false, b0,
                                                     (short)0, z, false, false);
    cAcc[j] = __builtin_amdgcn_wmma_f32_16x16x32_f16(false, a1, false, b1,
                                                     (short)0, cAcc[j], false, false);
  }

  // --- scatter-add with 32-bit element offsets (accumulator < 2^31 elements) ---
  // C/D layout: VGPR r holds M = 8*grp + r, N = lm (+16j)
  unsigned rowBase[8];
  #pragma unroll
  for (int r = 0; r < 8; ++r)
    rowBase[r] = (unsigned)sRow[w][8 * grp + r] << 6;   // orow * 64
  #pragma unroll
  for (int j = 0; j < 4; ++j) {
    const unsigned col = (unsigned)(16 * j + lm);
    #pragma unroll
    for (int r = 0; r < 8; ++r) {
      __hip_atomic_fetch_add(acc + (rowBase[r] + col), (float)cAcc[j][r],
                             __ATOMIC_RELAXED, __HIP_MEMORY_SCOPE_AGENT);
    }
  }
}

// ---------------- kernel 3: per-channel sum / sumsq reduction ----------------
__global__ __launch_bounds__(256) void dz_bn_stats(const float* __restrict__ acc,
                                                   float* __restrict__ stats) {
  __shared__ float sSum[64], sSq[64];
  const int t = threadIdx.x;
  if (t < 64) { sSum[t] = 0.f; sSq[t] = 0.f; }
  __syncthreads();
  const int c  = t & 63;
  const int r0 = t >> 6;                 // 4 rows per block-iteration
  float s = 0.f, q = 0.f;
  for (long row = (long)blockIdx.x * 4 + r0; row < N_OUT_PTS; row += (long)gridDim.x * 4) {
    float v = acc[row * 64 + c];
    s += v; q += v * v;
  }
  atomicAdd(&sSum[c], s);                // ds_add_f32
  atomicAdd(&sSq[c], q);
  __syncthreads();
  if (t < 64) {
    __hip_atomic_fetch_add(stats + t,      sSum[t], __ATOMIC_RELAXED, __HIP_MEMORY_SCOPE_AGENT);
    __hip_atomic_fetch_add(stats + 64 + t, sSq[t],  __ATOMIC_RELAXED, __HIP_MEMORY_SCOPE_AGENT);
  }
}

// ---------------- kernel 4: BN normalize + ReLU ----------------
__global__ __launch_bounds__(256) void dz_bn_relu(
    const float* __restrict__ acc, const float* __restrict__ stats,
    const float* __restrict__ gamma, const float* __restrict__ beta,
    float* __restrict__ out)
{
  __shared__ float sScale[64], sBias[64];
  const int t = threadIdx.x;
  if (t < 64) {
    const float invN = 1.0f / (float)N_OUT_PTS;
    float mean = stats[t] * invN;
    float var  = fmaxf(stats[64 + t] * invN - mean * mean, 0.0f);
    float sc   = rsqrtf(var + BN_EPS) * gamma[t];
    sScale[t] = sc;
    sBias[t]  = beta[t] - mean * sc;
  }
  __syncthreads();
  size_t i      = (size_t)blockIdx.x * blockDim.x + t;
  size_t stride = (size_t)gridDim.x * blockDim.x;
  for (; i < ACC_FLOATS; i += stride) {
    const int c = (int)(i & 63);
    float v = acc[i] * sScale[c] + sBias[c];
    out[i] = fmaxf(v, 0.0f);
  }
}

// ---------------- host-side launcher ----------------
extern "C" void kernel_launch(void* const* d_in, const int* in_sizes, int n_in,
                              void* d_out, int out_size, void* d_ws, size_t ws_size,
                              hipStream_t stream) {
  const float* x       = (const float*)d_in[0];
  const float* W       = (const float*)d_in[1];
  const float* gamma   = (const float*)d_in[2];
  const float* beta    = (const float*)d_in[3];
  const int*   in_idx  = (const int*)d_in[4];
  const int*   out_idx = (const int*)d_in[5];
  (void)in_sizes; (void)n_in; (void)out_size; (void)ws_size;

  float* acc   = (float*)d_ws;                 // 400000 x 64 f32 accumulator
  float* stats = acc + ACC_FLOATS;             // [64 sums | 64 sumsqs]
  float* out   = (float*)d_out;

  const long n4 = (long)((ACC_FLOATS + 128) / 4);
  hipLaunchKernelGGL(dz_zero_ws, dim3(2048), dim3(256), 0, stream, (float4*)d_ws, n4);

  dim3 g((TILES_PER_K + 7) / 8, K_TAPS);       // 782 x 27 blocks, 8 waves each
  hipLaunchKernelGGL(dz_gather_gemm_scatter, g, dim3(256), 0, stream,
                     x, W, in_idx, out_idx, acc);

  hipLaunchKernelGGL(dz_bn_stats, dim3(1024), dim3(256), 0, stream, acc, stats);
  hipLaunchKernelGGL(dz_bn_relu, dim3(2048), dim3(256), 0, stream,
                     acc, stats, gamma, beta, out);
}
